// Attention_6167573037579
// MI455X (gfx1250) — matrix operand order repaired and checked by reading, not compiled, not run on hardware
//
#include <hip/hip_runtime.h>
#include <hip/hip_bf16.h>
#include <stdint.h>

// ---------------- problem constants (from reference) ----------------
#define BATCH 4
#define SEQ   2048
#define DMODEL 512
#define NHEADS 8
#define DHEAD  64              // DMODEL / NHEADS
#define SCALE  0.125f          // 1/sqrt(64)
#define LOG2E  1.44269504088896340736f

typedef __attribute__((ext_vector_type(16))) __bf16 v16bf;
typedef __attribute__((ext_vector_type(8)))  __bf16 v8bf;
typedef __attribute__((ext_vector_type(8)))  float  v8f;
typedef __attribute__((ext_vector_type(4)))  float  v4f;

static __device__ __forceinline__ unsigned short f2bfu(float f) {
    unsigned u = __builtin_bit_cast(unsigned, f);
    unsigned r = u + 0x7FFFu + ((u >> 16) & 1u);   // round-to-nearest-even
    return (unsigned short)(r >> 16);
}

// alias-safe 16-byte load of 8 bf16 values (global or LDS)
static __device__ __forceinline__ v8bf ldb8(const unsigned short* p) {
    v8bf r;
    __builtin_memcpy(&r, p, 16);
    return r;
}

static __device__ __forceinline__ v16bf mk16(v8bf lo, v8bf hi) {
    return __builtin_shufflevector(lo, hi, 0,1,2,3,4,5,6,7,8,9,10,11,12,13,14,15);
}

static __device__ __forceinline__ v8f wmma_bf16(v16bf a, v16bf b, v8f c) {
    // D(f32 16x16) = A(bf16 16x32) * B(bf16 32x16) + C
    return __builtin_amdgcn_wmma_f32_16x16x32_bf16(
        /*neg_a=*/false, a, /*neg_b=*/false, b,
        /*c_mod=*/(short)0, c, /*reuse_a=*/false, /*reuse_b=*/false);
}

// async copy 16B global -> LDS (ASYNCcnt-tracked, ISA §10 VGLOBAL async)
static __device__ __forceinline__ void async_cp16(const unsigned short* g,
                                                  const unsigned short* l) {
    // generic LDS address: low 32 bits are the LDS byte offset (ISA §10.2)
    unsigned lds_off = (unsigned)(size_t)(const void*)l;
    asm volatile("global_load_async_to_lds_b128 %0, %1, off"
                 :: "v"(lds_off), "v"(g) : "memory");
}
static __device__ __forceinline__ void wait_async0() {
    asm volatile("s_wait_asynccnt 0x0" ::: "memory");
}

// ---------------- kernel 1: f32 -> bf16 convert (4-wide) ----------------
__global__ void k_cvt_bf16(const float* __restrict__ in,
                           unsigned short* __restrict__ out, int n4) {
    int i = blockIdx.x * blockDim.x + threadIdx.x;
    if (i < n4) {
        v4f v;
        __builtin_memcpy(&v, in + 4 * (size_t)i, 16);
        unsigned short o[4];
        o[0] = f2bfu(v[0]); o[1] = f2bfu(v[1]);
        o[2] = f2bfu(v[2]); o[3] = f2bfu(v[3]);
        __builtin_memcpy(out + 4 * (size_t)i, o, 8);
    }
}

// ---------------- kernel 2: LDS-staged bf16 GEMM  C = A * W^T + bias ----------
// A: [M][512] bf16 row-major, W: [512][512] bf16 row-major (N x K)
// Block = 8 waves = 128x128 tile. Per K-chunk of 32: the block cooperatively
// async-copies A[128x32] and W[128x32] into LDS (double-buffered, ASYNCcnt),
// then each wave runs 8 WMMAs on a 32x64 sub-tile from ds_load_b128 fragments.
// mode 0: out bf16 [B,H,S,Dh] (Q/K); mode 2: bf16 [B,H,Dh,S] (V^T);
// mode 3: out f32 row-major [M][512] (final projection)
__global__ __launch_bounds__(256, 1)
void k_gemm_bf16(const unsigned short* __restrict__ A,
                 const unsigned short* __restrict__ W,
                 const float* __restrict__ bias,
                 void* __restrict__ out, int mode) {
    __shared__ unsigned short Atile[2][128][32];   // 16 KB
    __shared__ unsigned short Wtile[2][128][32];   // 16 KB

    const int K = DMODEL, N = DMODEL;
    const int lane = threadIdx.x & 31;
    const int w    = threadIdx.x >> 5;
    const int half = lane >> 4;
    const int col  = lane & 15;
    const int mBase = blockIdx.x * 128;
    const int nBase = blockIdx.y * 128;
    const int ra0 = (w & 3) * 32;        // wave's A-row offset within tile
    const int rw0 = (w >> 2) * 64;       // wave's W-row offset within tile

    v8f acc[2][4];
    #pragma unroll
    for (int i = 0; i < 2; ++i)
        #pragma unroll
        for (int j = 0; j < 4; ++j) acc[i][j] = 0.f;

    // cooperative async staging of one 128x32 A tile + 128x32 W tile
    auto issue = [&](int k0, int buf) {
        const int l = threadIdx.x;
        #pragma unroll
        for (int u = 0; u < 2; ++u) {
            const int lin = l + 256 * u;          // 0..511 -> 512 x 16B segs
            const int row = lin >> 2;
            const int seg = (lin & 3) * 8;
            async_cp16(A + (size_t)(mBase + row) * K + k0 + seg,
                       &Atile[buf][row][seg]);
            async_cp16(W + (size_t)(nBase + row) * K + k0 + seg,
                       &Wtile[buf][row][seg]);
        }
    };

    issue(0, 0);
    wait_async0();
    __syncthreads();

    int buf = 0;
    for (int k0 = 0; k0 < K; k0 += 32) {
        if (k0 + 32 < K) issue(k0 + 32, buf ^ 1);  // overlap DMA with WMMAs

        v16bf a[2];
        #pragma unroll
        for (int i = 0; i < 2; ++i) {
            const unsigned short* ar = &Atile[buf][ra0 + i * 16 + col][0];
            a[i] = mk16(ldb8(ar + 8 * half), ldb8(ar + 16 + 8 * half));
        }
        v16bf b[4];
        #pragma unroll
        for (int j = 0; j < 4; ++j) {
            const unsigned short* wr = &Wtile[buf][rw0 + j * 16 + col][0];
            b[j] = mk16(ldb8(wr + 8 * half), ldb8(wr + 16 + 8 * half));
        }
        #pragma unroll
        for (int i = 0; i < 2; ++i)
            #pragma unroll
            for (int j = 0; j < 4; ++j)
                acc[i][j] = wmma_bf16(a[i], b[j], acc[i][j]);

        wait_async0();      // next chunk landed in LDS
        __syncthreads();    // visible to all waves; old buffer free
        buf ^= 1;
    }

    #pragma unroll
    for (int i = 0; i < 2; ++i)
      #pragma unroll
      for (int j = 0; j < 4; ++j) {
        const int n = nBase + rw0 + j * 16 + col;
        const float bn = bias[n];
        #pragma unroll
        for (int r = 0; r < 8; ++r) {
            const int m = mBase + ra0 + i * 16 + 8 * half + r;
            const float v = acc[i][j][r] + bn;
            if (mode == 3) {
                ((float*)out)[(size_t)m * N + n] = v;
            } else {
                const int b_ = m >> 11, s = m & (SEQ - 1);  // m = b*SEQ + s
                const int h = n >> 6, dh = n & 63;          // n = h*64 + dh
                size_t idx;
                if (mode == 0)  // [B,H,S,Dh]
                    idx = ((size_t)((b_ * NHEADS + h) * SEQ + s)) * DHEAD + dh;
                else            // mode 2: [B,H,Dh,S]
                    idx = ((size_t)((b_ * NHEADS + h) * DHEAD + dh)) * SEQ + s;
                ((unsigned short*)out)[idx] = f2bfu(v);
            }
        }
      }
}

// ---------------- kernel 3: causal flash attention ----------------
// Q,K: [B,H,S,Dh] bf16; Vt: [B,H,Dh,S] bf16; O: [B,S,D] bf16
// One wave owns a 32-query tile (two 16-row fragments sharing K/V fragments);
// iterates keys in chunks of 32 with online softmax. 16 WMMAs per key chunk.
__global__ __launch_bounds__(256, 1)
void k_flash_attn(const unsigned short* __restrict__ Q,
                  const unsigned short* __restrict__ Kh,
                  const unsigned short* __restrict__ Vt,
                  unsigned short* __restrict__ O) {
    __shared__ unsigned short ptile[8][2][16][32];  // per-wave 2x(16x32) P staging

    const int lane = threadIdx.x & 31;
    const int w    = threadIdx.x >> 5;
    const int half = lane >> 4;
    const int col  = lane & 15;

    const int tile = blockIdx.x * 8 + w;            // 0 .. B*H*(S/32)-1
    const int qt = tile & (SEQ / 32 - 1);
    const int bh = tile >> 6;                       // b*H + h
    const int q0 = qt * 32;

    const unsigned short* Qb = Q  + (size_t)bh * SEQ * DHEAD;
    const unsigned short* Kb = Kh + (size_t)bh * SEQ * DHEAD;
    const unsigned short* Vb = Vt + (size_t)bh * DHEAD * SEQ;

    // Q fragments: 2 row tiles x 2 Dh-chunks
    v16bf qf[2][2];
    #pragma unroll
    for (int i = 0; i < 2; ++i) {
        const unsigned short* Qr = Qb + (size_t)(q0 + i * 16 + col) * DHEAD;
        qf[i][0] = mk16(ldb8(Qr + 0  + 8 * half), ldb8(Qr + 16 + 8 * half));
        qf[i][1] = mk16(ldb8(Qr + 32 + 8 * half), ldb8(Qr + 48 + 8 * half));
    }

    v8f o[2][4];
    float mrow[2][8], lrow[2][8];
    #pragma unroll
    for (int i = 0; i < 2; ++i) {
        #pragma unroll
        for (int t = 0; t < 4; ++t) o[i][t] = 0.f;
        #pragma unroll
        for (int r = 0; r < 8; ++r) { mrow[i][r] = -1e30f; lrow[i][r] = 0.f; }
    }

    for (int kb = 0; kb <= q0; kb += 32) {          // causal: keys <= q0+31
        // ---- scores: S[32x32] = Q[32x64] * K^T ----
        v8f s[2][2];
        #pragma unroll
        for (int i = 0; i < 2; ++i) { s[i][0] = 0.f; s[i][1] = 0.f; }
        #pragma unroll
        for (int dk = 0; dk < 2; ++dk) {
            const int k0d = dk * 32;
            const unsigned short* kr0 =
                Kb + (size_t)(kb + col) * DHEAD + k0d + 8 * half;
            const unsigned short* kr1 =
                Kb + (size_t)(kb + 16 + col) * DHEAD + k0d + 8 * half;
            v16bf b0 = mk16(ldb8(kr0), ldb8(kr0 + 16));
            v16bf b1 = mk16(ldb8(kr1), ldb8(kr1 + 16));
            #pragma unroll
            for (int i = 0; i < 2; ++i) {
                s[i][0] = wmma_bf16(qf[i][dk], b0, s[i][0]);
                s[i][1] = wmma_bf16(qf[i][dk], b1, s[i][1]);
            }
        }

        // ---- online softmax (per row tile) + P staging ----
        v16bf pf[2];
        #pragma unroll
        for (int i = 0; i < 2; ++i) {
            float p0[8], p1[8], alpha[8];
            #pragma unroll
            for (int r = 0; r < 8; ++r) {
                const int q = q0 + i * 16 + 8 * half + r;
                float v0 = s[i][0][r] * SCALE; if (kb + col      > q) v0 = -1e30f;
                float v1 = s[i][1][r] * SCALE; if (kb + 16 + col > q) v1 = -1e30f;
                float mx = fmaxf(v0, v1);
                #pragma unroll
                for (int off = 8; off; off >>= 1)
                    mx = fmaxf(mx, __shfl_xor(mx, off, 32));  // 16-lane group
                const float mn = fmaxf(mrow[i][r], mx);
                const float a  = exp2f((mrow[i][r] - mn) * LOG2E);
                p0[r] = exp2f((v0 - mn) * LOG2E);
                p1[r] = exp2f((v1 - mn) * LOG2E);
                float rs = p0[r] + p1[r];
                #pragma unroll
                for (int off = 8; off; off >>= 1)
                    rs += __shfl_xor(rs, off, 32);
                mrow[i][r] = mn;
                lrow[i][r] = lrow[i][r] * a + rs;
                alpha[r] = a;
            }
            #pragma unroll
            for (int t = 0; t < 4; ++t)
                #pragma unroll
                for (int r = 0; r < 8; ++r)
                    o[i][t][r] *= alpha[r];
            // repack P: C-layout -> A-layout via per-wave LDS tile
            #pragma unroll
            for (int r = 0; r < 8; ++r) {
                ptile[w][i][8 * half + r][col]      = f2bfu(p0[r]);
                ptile[w][i][8 * half + r][col + 16] = f2bfu(p1[r]);
            }
            const unsigned short* pr = &ptile[w][i][col][0];  // A row = lane&15
            pf[i] = mk16(ldb8(pr + 8 * half), ldb8(pr + 16 + 8 * half));
        }

        // ---- O[32x64] += P[32x32] * V[32x64]  (V stored transposed) ----
        #pragma unroll
        for (int t = 0; t < 4; ++t) {
            const unsigned short* vr =
                Vb + (size_t)(t * 16 + col) * SEQ + kb + 8 * half;
            v16bf vfrag = mk16(ldb8(vr), ldb8(vr + 16));
            o[0][t] = wmma_bf16(pf[0], vfrag, o[0][t]);
            o[1][t] = wmma_bf16(pf[1], vfrag, o[1][t]);
        }
    }

    // ---- epilogue: O / l, write bf16 [B,S,D] ----
    const int b = bh >> 3, h = bh & 7;
    #pragma unroll
    for (int i = 0; i < 2; ++i)
        #pragma unroll
        for (int t = 0; t < 4; ++t)
            #pragma unroll
            for (int r = 0; r < 8; ++r) {
                const int q = q0 + i * 16 + 8 * half + r;
                const float val = o[i][t][r] / lrow[i][r];
                O[((size_t)(b * SEQ + q)) * DMODEL + h * DHEAD + t * 16 + col] =
                    f2bfu(val);
            }
}

// ---------------- host-side orchestration ----------------
extern "C" void kernel_launch(void* const* d_in, const int* in_sizes, int n_in,
                              void* d_out, int out_size, void* d_ws, size_t ws_size,
                              hipStream_t stream) {
    (void)in_sizes; (void)n_in; (void)out_size; (void)ws_size;
    const float* x   = (const float*)d_in[0];
    // d_in[1] = mask (causal, recomputed analytically)
    const float* Wq  = (const float*)d_in[2];
    const float* bq  = (const float*)d_in[3];
    const float* Wk  = (const float*)d_in[4];
    const float* bk  = (const float*)d_in[5];
    const float* Wv  = (const float*)d_in[6];
    const float* bv  = (const float*)d_in[7];
    const float* Wo  = (const float*)d_in[8];
    const float* bo  = (const float*)d_in[9];

    const int M  = BATCH * SEQ;            // 8192
    const int NX = M * DMODEL;             // 4,194,304
    const int NW = DMODEL * DMODEL;        // 262,144

    // workspace layout (bf16 = ushort), all 16B aligned
    char* ws = (char*)d_ws;
    unsigned short* xb  = (unsigned short*)(ws);
    unsigned short* Wqb = (unsigned short*)(ws + (size_t)NX * 2);
    unsigned short* Wkb = Wqb + NW;
    unsigned short* Wvb = Wkb + NW;
    unsigned short* Wob = Wvb + NW;
    unsigned short* Qh  = Wob + NW;                                  // [B,H,S,Dh]
    unsigned short* Kh  = Qh + (size_t)NX;
    unsigned short* Vt  = Kh + (size_t)NX;                           // [B,H,Dh,S]
    unsigned short* Ob  = Vt + (size_t)NX;                           // [B,S,D]

    // 1) convert to bf16 (4 elems/thread)
    k_cvt_bf16<<<(NX / 4 + 255) / 256, 256, 0, stream>>>(x,  xb,  NX / 4);
    k_cvt_bf16<<<(NW / 4 + 255) / 256, 256, 0, stream>>>(Wq, Wqb, NW / 4);
    k_cvt_bf16<<<(NW / 4 + 255) / 256, 256, 0, stream>>>(Wk, Wkb, NW / 4);
    k_cvt_bf16<<<(NW / 4 + 255) / 256, 256, 0, stream>>>(Wv, Wvb, NW / 4);
    k_cvt_bf16<<<(NW / 4 + 255) / 256, 256, 0, stream>>>(Wo, Wob, NW / 4);

    // 2) Q/K/V projections (async-LDS staged WMMA GEMM), 128x128 block tiles
    dim3 ggrid(M / 128, DMODEL / 128);
    k_gemm_bf16<<<ggrid, 256, 0, stream>>>(xb, Wqb, bq, (void*)Qh, 0);
    k_gemm_bf16<<<ggrid, 256, 0, stream>>>(xb, Wkb, bk, (void*)Kh, 0);
    k_gemm_bf16<<<ggrid, 256, 0, stream>>>(xb, Wvb, bv, (void*)Vt, 2);

    // 3) causal flash attention: B*H*(S/32) = 2048 wave-tiles, 8 waves/block
    k_flash_attn<<<(BATCH * NHEADS * (SEQ / 32)) / 8, 256, 0, stream>>>(
        Qh, Kh, Vt, Ob);

    // 4) output projection -> f32 d_out
    k_gemm_bf16<<<ggrid, 256, 0, stream>>>(Ob, Wob, bo, d_out, 3);
}
